// QuantumLSTMClassifier_53214644797981
// MI455X (gfx1250) — matrix operand order repaired and checked by reading, not compiled, not run on hardware
//
#include <hip/hip_runtime.h>
#include <hip/hip_bf16.h>
#include <math.h>

typedef _Float16 f16;
typedef __attribute__((ext_vector_type(16))) _Float16 v16h;
typedef __attribute__((ext_vector_type(8)))  _Float16 v8h;
typedef __attribute__((ext_vector_type(4)))  _Float16 v4h;
typedef __attribute__((ext_vector_type(8)))  float    v8f;

// Problem sizes (fixed by setup_inputs)
// B=1024, T=128, E=128, H=256, NQ=8, C=2

// ---------------------------------------------------------------------------
// K1: constant gate values per hidden unit + LSTM scan.
// q_feat is input-independent (RX embedding on |+> is a global phase):
// z_w = -sin(qw[g][0][w]); qf[0] = prod_{w=1..7} z_w ; qf[k] = prod_{w=0..k}.
// f,i,g,o constant per h -> h_seq(t,h) = o * tanh(c_t), c_t = f*c + i*g.
// One block, 256 threads = one thread per hidden unit.
// ---------------------------------------------------------------------------
__global__ __launch_bounds__(256) void k_gates_scan(
    const float* __restrict__ qw,    // (4,2,8)
    const float* __restrict__ Wo,    // (4,256,8)
    const float* __restrict__ bo,    // (4,256)
    float* __restrict__ hseq)        // (128,256) f32, workspace
{
    const int h = threadIdx.x;
    float G[4];
#pragma unroll
    for (int g = 0; g < 4; ++g) {
        float z[8];
#pragma unroll
        for (int w = 0; w < 8; ++w) z[w] = -sinf(qw[g * 16 + w]); // qw[g][0][w]
        float qf[8];
        float p = 1.0f;
#pragma unroll
        for (int w = 1; w < 8; ++w) p *= z[w];
        qf[0] = p;                      // Z_0 = z1*...*z7
        p = z[0];
#pragma unroll
        for (int k = 1; k < 8; ++k) { p *= z[k]; qf[k] = p; } // Z_k = z0*...*zk
        float acc = bo[g * 256 + h];
        const float* wr = Wo + (g * 256 + h) * 8;
#pragma unroll
        for (int q = 0; q < 8; ++q) acc += wr[q] * qf[q];
        G[g] = acc;
    }
    const float f  = 1.0f / (1.0f + expf(-G[0]));
    const float ii = 1.0f / (1.0f + expf(-G[1]));
    const float gg = tanhf(G[2]);
    const float o  = 1.0f / (1.0f + expf(-G[3]));
    const float ig = ii * gg;
    float c = 0.0f;
    for (int t = 0; t < 128; ++t) {
        c = f * c + ig;
        hseq[t * 256 + h] = o * tanhf(c);
    }
}

// ---------------------------------------------------------------------------
// K2: fold the classifier head through the pooling sum and emit the WMMA
// B fragments PRE-SWIZZLED into the 16x16x32-f16 lane layout.
// P[t][c] = sum_h hseq[t][h] * fcW[c][h]; logical B (128x16):
//   col0 = P0, col1 = P1, col2 = 1.0 (count), cols 3..15 = 0.
// B-fragment layout (ISA 7.12.2): step kk, lane L -> N = L&15,
//   element e -> K = (e<8 ? 0 : 16) + (L<16 ? 0 : 8) + (e&7), t = kk*32 + K.
// Stored flat as Bfrag[((kk*32 + lane)*16 + e)] so K4 does one aligned
// 32-byte vector load per K-step. 256 threads: phase1 (t,c) dots -> LDS,
// phase2 each thread writes 8 contiguous swizzled f16 (16-byte store).
// ---------------------------------------------------------------------------
__global__ __launch_bounds__(256) void k_project(
    const float* __restrict__ hseq,  // (128,256)
    const float* __restrict__ fcW,   // (2,256)
    f16* __restrict__ Bfrag)         // (4*32*16) f16, workspace
{
    __shared__ float Pl[128 * 2];    // [t][c]
    const int idx = threadIdx.x;
    const int t = idx >> 1;
    const int c = idx & 1;
    const float* hr = hseq + t * 256;
    const float* wr = fcW + c * 256;
    float s = 0.0f;
    for (int h = 0; h < 256; ++h) s += hr[h] * wr[h];
    Pl[t * 2 + c] = s;
    __syncthreads();

    const int base = idx * 8;        // 2048 entries / 256 threads
    v8h vals;
#pragma unroll
    for (int j = 0; j < 8; ++j) {
        const int i    = base + j;
        const int kk   = i >> 9;          // K-step
        const int lane = (i >> 4) & 31;   // destination lane
        const int e    = i & 15;          // element within lane
        const int tt   = kk * 32 + ((e < 8) ? 0 : 16) + ((lane < 16) ? 0 : 8) + (e & 7);
        const int col  = lane & 15;
        float v;
        if      (col == 0) v = Pl[tt * 2 + 0];
        else if (col == 1) v = Pl[tt * 2 + 1];
        else if (col == 2) v = 1.0f;      // mask-count column
        else               v = 0.0f;
        vals[j] = (f16)v;
    }
    *(v8h*)(Bfrag + base) = vals;
}

// ---------------------------------------------------------------------------
// K3: token ids -> f16 mask (exact 0.0 / 1.0). int4 load + 8B store, coalesced.
// ---------------------------------------------------------------------------
__global__ __launch_bounds__(256) void k_mask(
    const int* __restrict__ x, f16* __restrict__ m)
{
    const int i = blockIdx.x * 256 + threadIdx.x;  // grid covers 1024*128/4
    const int4 xv = ((const int4*)x)[i];
    v4h mv;
    mv[0] = (xv.x != 0) ? (f16)1.0f : (f16)0.0f;
    mv[1] = (xv.y != 0) ? (f16)1.0f : (f16)0.0f;
    mv[2] = (xv.z != 0) ? (f16)1.0f : (f16)0.0f;
    mv[3] = (xv.w != 0) ? (f16)1.0f : (f16)0.0f;
    ((v4h*)m)[i] = mv;
}

// ---------------------------------------------------------------------------
// K4: WMMA masked-pool + head: C(16x16) = mask_tile(16x128) @ B(128x16).
// A layout (f16 16x32): lane L -> M = L&15; lanes 0-15 hold K=0..7 (elems
// 0..7) and K=16..23 (elems 8..15); lanes 16-31 hold K=8..15 and 24..31.
// B comes pre-swizzled from K2 (one 32B vector load per K-step).
// D: VGPR r -> M = r + (lane>=16 ? 8 : 0), N = lane&15; columns {0,1,2}
// of D are {sum mask*P0, sum mask*P1, count}.
// 4 waves per block, one 16-row batch tile per wave; 4 K-steps of K=32.
// ---------------------------------------------------------------------------
__global__ __launch_bounds__(128) void k_pool_wmma(
    const f16* __restrict__ mask,    // (1024,128)
    const f16* __restrict__ Bfrag,   // (4*32*16)
    const float* __restrict__ fcb,   // (2,)
    float* __restrict__ out)         // (1024,2)
{
    const int lane   = threadIdx.x & 31;
    const int wid    = threadIdx.x >> 5;
    const int tile   = blockIdx.x * 4 + wid;   // 16 blocks x 4 waves = 64 tiles
    const int mlocal = lane & 15;
    const int hi     = lane >> 4;
    const int kbaseA = hi ? 8 : 0;
    const int mrow   = tile * 16 + mlocal;

    v8f acc = {};
#pragma unroll
    for (int kk = 0; kk < 4; ++kk) {
        const f16* arow = mask + mrow * 128 + kk * 32 + kbaseA;
        v8h a0 = *(const v8h*)(arow);        // K = kbaseA + 0..7
        v8h a1 = *(const v8h*)(arow + 16);   // K = kbaseA + 16..23
        v16h a = __builtin_shufflevector(a0, a1,
                 0, 1, 2, 3, 4, 5, 6, 7, 8, 9, 10, 11, 12, 13, 14, 15);
        v16h b = *(const v16h*)(Bfrag + (kk * 32 + lane) * 16);
        acc = __builtin_amdgcn_wmma_f32_16x16x32_f16(
            /*neg_a=*/false, a, /*neg_b=*/false, b,
            /*c_mod=*/(short)0, acc, /*reuse_a=*/false, /*reuse_b=*/false);
    }

    __shared__ float s[4][16 * 3];
    if (mlocal < 3) {
#pragma unroll
        for (int r = 0; r < 8; ++r) s[wid][(r + hi * 8) * 3 + mlocal] = acc[r];
    }
    __syncthreads();

    const int brow = lane >> 1;      // 16 rows x 2 classes = 32 lanes
    const int c    = lane & 1;
    const float num = s[wid][brow * 3 + c];
    const float cnt = s[wid][brow * 3 + 2];
    out[(tile * 16 + brow) * 2 + c] = num / (cnt + 1e-9f) + fcb[c];
}

// ---------------------------------------------------------------------------
// Launch. Inputs: 0:x 1:embed 2:Wi 3:bi 4:qw 5:Wo 6:bo 7:fcW 8:fcb.
// embed/Wi/bi are provably dead (RX embedding on |+> is a global phase).
// ---------------------------------------------------------------------------
extern "C" void kernel_launch(void* const* d_in, const int* in_sizes, int n_in,
                              void* d_out, int out_size, void* d_ws, size_t ws_size,
                              hipStream_t stream) {
    (void)in_sizes; (void)n_in; (void)out_size; (void)ws_size;
    const int*   x   = (const int*)  d_in[0];
    const float* qw  = (const float*)d_in[4];
    const float* Wo  = (const float*)d_in[5];
    const float* bo  = (const float*)d_in[6];
    const float* fcW = (const float*)d_in[7];
    const float* fcb = (const float*)d_in[8];
    float* out = (float*)d_out;

    char* ws = (char*)d_ws;
    float* hseq  = (float*)ws;                       // 128*256*4   = 131072 B
    f16*   Bfrag = (f16*)(ws + 131072);              // 4*32*16*2   =   4096 B
    f16*   mskh  = (f16*)(ws + 131072 + 4096);       // 1024*128*2  = 262144 B

    k_gates_scan<<<1,   256, 0, stream>>>(qw, Wo, bo, hseq);
    k_project  <<<1,   256, 0, stream>>>(hseq, fcW, Bfrag);
    k_mask     <<<128, 256, 0, stream>>>(x, mskh);
    k_pool_wmma<<<16,  128, 0, stream>>>(mskh, Bfrag, fcb, out);
}
